// block_bone_input_18743237280151
// MI455X (gfx1250) — matrix operand order repaired and checked by reading, not compiled, not run on hardware
//
#include <hip/hip_runtime.h>
#include <hip/hip_bf16.h>

typedef __attribute__((ext_vector_type(16))) _Float16 v16h;
typedef __attribute__((ext_vector_type(8)))  float    v8f;

constexpr int kV  = 25;
constexpr int kC  = 3;
constexpr int kT  = 300;
constexpr int kN  = 256;
constexpr int kCH = 75;            // C*V
constexpr int kNT = kN * kT;       // 76800 samples per BN channel
constexpr int kP  = kV * kT;       // 7500 positions per sample
constexpr int kTileP = 128;        // positions per block

// parent[v1-1] = v2-1 from ntu pairs
__constant__ int c_parent[kV] = {1,20,20,2,20,4,5,6,20,8,9,10,0,12,13,14,0,16,17,18,20,22,7,24,11};

// ---- workspace layout (float offsets) ----
// acc:    [0,300)   sum_bone(75) sumsq_bone(75) sum_tb(75) sumsq_tb(75)
// Weff1:  [320, 5120)    4800
// Beff1:  [5120, 6720)   1600
// Weff2:  [6720, 11520)  4800
// Beff2:  [11520,13120)  1600
// Wh f16: starting at float offset 13120: 8192 halves (w1b then w2b)
#define OFF_ACC   0
#define OFF_WEFF1 320
#define OFF_BEFF1 5120
#define OFF_WEFF2 6720
#define OFF_BEFF2 11520
#define OFF_WH_F  13120

union H16 { v16h v; _Float16 h[16]; };
union F8  { v8f  v; float    f[8];  };

__global__ void k_zero(float* ws) {
    int i = threadIdx.x;
    if (i < 300) ws[OFF_ACC + i] = 0.0f;
}

// One block per sample n; threads 0..74 own one (v,c) channel, march over T.
__global__ void k_stats(const float* __restrict__ x, float* __restrict__ ws) {
    int n  = blockIdx.x;
    int ch = threadIdx.x;
    if (ch >= kCH) return;
    int v = ch / kC, c = ch % kC;
    int pv = c_parent[v];
    const float* xb = x + (size_t)n * (kT * kCH);
    float sb = 0.f, qb = 0.f, st = 0.f, qt = 0.f, prev = 0.f;
    for (int t = 0; t < kT; ++t) {
        const float* r = xb + t * kCH;
        float b = r[v * kC + c] - r[pv * kC + c];
        sb += b; qb += b * b;
        if (t > 0) { float tb = b - prev; st += tb; qt += tb * tb; }
        prev = b;
    }
    // t_bone last frame is zero-padded: contributes 0 to sums, denominator stays N*T.
    int chbn = c * kV + v;   // BN channel order = c*V + v
    atomicAdd(&ws[OFF_ACC +       chbn], sb);
    atomicAdd(&ws[OFF_ACC +  75 + chbn], qb);
    atomicAdd(&ws[OFF_ACC + 150 + chbn], st);
    atomicAdd(&ws[OFF_ACC + 225 + chbn], qt);
}

// Finalize BN stats, fold BN + 3->64 conv into Weff/Beff, convert 64x64 weights to f16.
__global__ void k_pre(const float* __restrict__ g1, const float* __restrict__ bb1,
                      const float* __restrict__ w1a, const float* __restrict__ b1a,
                      const float* __restrict__ w1b,
                      const float* __restrict__ g2, const float* __restrict__ bb2,
                      const float* __restrict__ w2a, const float* __restrict__ b2a,
                      const float* __restrict__ w2b,
                      float* __restrict__ ws) {
    __shared__ float s1[kCH], t1[kCH], s2[kCH], t2[kCH];
    int tid = threadIdx.x;
    if (tid < kCH) {
        const float inv = 1.0f / (float)kNT;
        float m   = ws[OFF_ACC + tid] * inv;
        float var = ws[OFF_ACC + 75 + tid] * inv - m * m;
        float sc  = g1[tid] * rsqrtf(var + 1e-5f);
        s1[tid] = sc; t1[tid] = bb1[tid] - m * sc;
        m   = ws[OFF_ACC + 150 + tid] * inv;
        var = ws[OFF_ACC + 225 + tid] * inv - m * m;
        sc  = g2[tid] * rsqrtf(var + 1e-5f);
        s2[tid] = sc; t2[tid] = bb2[tid] - m * sc;
    }
    __syncthreads();
    // Weff[(o*3+c)*25+v] = wa[o,c] * scale[c*25+v]
    for (int i = tid; i < 4800; i += blockDim.x) {
        int v = i % 25; int rc = i / 25; int c = rc % 3;
        ws[OFF_WEFF1 + i] = w1a[rc] * s1[c * 25 + v];
        ws[OFF_WEFF2 + i] = w2a[rc] * s2[c * 25 + v];
    }
    // Beff[o*25+v] = ba[o] + sum_c wa[o,c]*shift[c*25+v]
    for (int i = tid; i < 1600; i += blockDim.x) {
        int v = i % 25; int o = i / 25;
        ws[OFF_BEFF1 + i] = b1a[o] + w1a[o*3+0]*t1[v] + w1a[o*3+1]*t1[25+v] + w1a[o*3+2]*t1[50+v];
        ws[OFF_BEFF2 + i] = b2a[o] + w2a[o*3+0]*t2[v] + w2a[o*3+1]*t2[25+v] + w2a[o*3+2]*t2[50+v];
    }
    _Float16* wh = (_Float16*)(ws + OFF_WH_F);
    for (int i = tid; i < 4096; i += blockDim.x) {
        wh[i]        = (_Float16)w1b[i];
        wh[4096 + i] = (_Float16)w2b[i];
    }
}

// Main fused kernel: one block = (sample n) x (128-position tile).
__global__ __launch_bounds__(256) void k_main(const float* __restrict__ x,
                                              const float* __restrict__ ws,
                                              const float* __restrict__ b1b,
                                              const float* __restrict__ b2b,
                                              float* __restrict__ out) {
    __shared__ _Float16 WhS[2 * 4096];            // both 64x64 f16 weight matrices, 16KB
    __shared__ _Float16 Hs[2 * kTileP * 64];      // H tiles [branch][pos][k], 32KB
    const int n   = blockIdx.y;
    const int p0  = blockIdx.x * kTileP;
    const int tid = threadIdx.x;

    // stage f16 weights to LDS (8192 halves = 4096 dwords, coalesced)
    {
        const unsigned int* src = (const unsigned int*)(ws + OFF_WH_F);
        unsigned int* dst = (unsigned int*)WhS;
        for (int i = tid; i < 4096; i += 256) dst[i] = src[i];
    }

    // ---- phase 1: bone/t_bone -> folded 3->64 layer -> H tiles in LDS ----
    {
        int j  = tid & 127;          // local position
        int oh = tid >> 7;           // output-channel half (0/1)
        int p  = p0 + j;
        bool valid = p < kP;
        int v = valid ? p / kT : 0;
        int t = valid ? p - v * kT : 0;
        int pv = c_parent[v];
        float b[3]  = {0.f, 0.f, 0.f};
        float tb[3] = {0.f, 0.f, 0.f};
        if (valid) {
            const float* r0 = x + ((size_t)n * kT + t) * kCH;
            #pragma unroll
            for (int c = 0; c < 3; ++c) b[c] = r0[v*3 + c] - r0[pv*3 + c];
            if (t < kT - 1) {
                const float* r1 = r0 + kCH;
                #pragma unroll
                for (int c = 0; c < 3; ++c) tb[c] = (r1[v*3 + c] - r1[pv*3 + c]) - b[c];
            }
        }
        const float* We1 = ws + OFF_WEFF1;
        const float* Be1 = ws + OFF_BEFF1;
        const float* We2 = ws + OFF_WEFF2;
        const float* Be2 = ws + OFF_BEFF2;
        for (int o = oh * 32; o < oh * 32 + 32; ++o) {
            float a1 = Be1[o*25 + v] + We1[(o*3+0)*25 + v]*b[0]
                                     + We1[(o*3+1)*25 + v]*b[1]
                                     + We1[(o*3+2)*25 + v]*b[2];
            float a2 = Be2[o*25 + v] + We2[(o*3+0)*25 + v]*tb[0]
                                     + We2[(o*3+1)*25 + v]*tb[1]
                                     + We2[(o*3+2)*25 + v]*tb[2];
            Hs[0*kTileP*64 + j*64 + o] = (_Float16)(valid ? fmaxf(a1, 0.f) : 0.f);
            Hs[1*kTileP*64 + j*64 + o] = (_Float16)(valid ? fmaxf(a2, 0.f) : 0.f);
        }
    }
    __syncthreads();

    // ---- phase 2: Y = relu(W1b@H1+b1b) + relu(W2b@H2+b2b) via v_wmma_f32_16x16x32_f16
    const int lane = tid & 31;
    const int hsel = lane >> 4;       // lane half (0/1)
    const int nc   = lane & 15;       // matrix row/col within half
    const int ct   = tid >> 5;        // wave id = column tile 0..7

    // B fragments (32x16): lane holds col N=nc; K = k0 + 16*hsel + e  (contiguous order, same as A)
    H16 bf[2][2];
    #pragma unroll
    for (int br = 0; br < 2; ++br)
        #pragma unroll
        for (int kk = 0; kk < 2; ++kk) {
            const _Float16* hb = &Hs[br*kTileP*64 + (ct*16 + nc)*64 + kk*32 + 16*hsel];
            #pragma unroll
            for (int e = 0; e < 16; ++e) bf[br][kk].h[e] = hb[e];
        }

    #pragma unroll
    for (int rt = 0; rt < 4; ++rt) {
        F8 acc1; acc1.v = (v8f){0,0,0,0,0,0,0,0};
        F8 acc2; acc2.v = (v8f){0,0,0,0,0,0,0,0};
        #pragma unroll
        for (int kk = 0; kk < 2; ++kk) {
            // A fragment (16x32): row M=nc; K = k0 + 16*hsel + e (contiguous order, same as B)
            H16 a1, a2;
            #pragma unroll
            for (int e = 0; e < 16; ++e) {
                int k  = kk*32 + 16*hsel + e;
                a1.h[e] = WhS[        (rt*16 + nc)*64 + k];
                a2.h[e] = WhS[4096 +  (rt*16 + nc)*64 + k];
            }
            acc1.v = __builtin_amdgcn_wmma_f32_16x16x32_f16(false, a1.v, false, bf[0][kk].v,
                                                            (short)0, acc1.v, false, false);
            acc2.v = __builtin_amdgcn_wmma_f32_16x16x32_f16(false, a2.v, false, bf[1][kk].v,
                                                            (short)0, acc2.v, false, false);
        }
        // epilogue: bias + relu per branch, sum, store fp32
        int pcol = p0 + ct*16 + nc;
        #pragma unroll
        for (int r = 0; r < 8; ++r) {
            int o = rt*16 + 8*hsel + r;   // C/D layout: VGPR r holds row M = r + 8*half
            float y = fmaxf(acc1.f[r] + b1b[o], 0.f) + fmaxf(acc2.f[r] + b2b[o], 0.f);
            if (pcol < kP)
                out[((size_t)n * 64 + o) * kP + pcol] = y;
        }
    }
}

extern "C" void kernel_launch(void* const* d_in, const int* in_sizes, int n_in,
                              void* d_out, int out_size, void* d_ws, size_t ws_size,
                              hipStream_t stream) {
    (void)in_sizes; (void)n_in; (void)out_size; (void)ws_size;
    const float* x    = (const float*)d_in[0];
    const float* g1   = (const float*)d_in[1];
    const float* bb1  = (const float*)d_in[2];
    const float* w1a  = (const float*)d_in[3];
    const float* b1a  = (const float*)d_in[4];
    const float* w1b  = (const float*)d_in[5];
    const float* b1b  = (const float*)d_in[6];
    const float* g2   = (const float*)d_in[7];
    const float* bb2  = (const float*)d_in[8];
    const float* w2a  = (const float*)d_in[9];
    const float* b2a  = (const float*)d_in[10];
    const float* w2b  = (const float*)d_in[11];
    const float* b2b  = (const float*)d_in[12];
    float* ws  = (float*)d_ws;
    float* out = (float*)d_out;

    k_zero<<<1, 320, 0, stream>>>(ws);
    k_stats<<<kN, 128, 0, stream>>>(x, ws);
    k_pre<<<1, 256, 0, stream>>>(g1, bb1, w1a, b1a, w1b, g2, bb2, w2a, b2a, w2b, ws);
    dim3 grid((kP + kTileP - 1) / kTileP, kN);   // 59 x 256
    k_main<<<grid, 256, 0, stream>>>(x, ws, b1b, b2b, out);
}
